// Codebook_73942156968166
// MI455X (gfx1250) — matrix operand order repaired and checked by reading, not compiled, not run on hardware
//
#include <hip/hip_runtime.h>

// ---------------------------------------------------------------------------
// VQ-VAE codebook quantization for MI455X (gfx1250, wave32, WMMA + TDM).
//
// d[n,k] = ||z_n||^2 + ||e_k||^2 - 2 z_n.e_k ; argmin needs ||e_k||^2 - 2 z.e.
// The 32768x1024x256 dot GEMM runs on v_wmma_f32_16x16x32_bf16 with an
// error-compensated bf16 hi/lo split (hi*hi + hi*lo + lo*hi ~ fp32 accuracy).
// B tiles (16 codes x 256 dims, bf16) are DMA'd into LDS by the Tensor Data
// Mover (tensor_load_to_lds, TENSORcnt), double-buffered so the DMA of tile
// t+1 overlaps the WMMAs of tile t; all 8 waves of the block share each tile.
// ---------------------------------------------------------------------------

typedef __attribute__((ext_vector_type(16))) __bf16          v16bf;
typedef __attribute__((ext_vector_type(16))) unsigned short  v16u;
typedef __attribute__((ext_vector_type(8)))  float           v8f;
typedef __attribute__((ext_vector_type(4)))  unsigned int    v4u;
typedef __attribute__((ext_vector_type(8)))  int             v8i;
typedef __attribute__((ext_vector_type(4)))  int             v4i;

#define K_CODES 1024
#define CDIM    256
#define N_VEC   32768          // B*H*W = 32*32*32
#define HWX     1024           // H*W
#define BETA_F  0.25f
#define OUTQ_ELEMS 8388608LL   // 32*256*32*32

__device__ __forceinline__ unsigned short bf16_rne(float f) {
    unsigned int u = __float_as_uint(f);
    u += 0x7FFFu + ((u >> 16) & 1u);      // round-to-nearest-even
    return (unsigned short)(u >> 16);
}
__device__ __forceinline__ float bf16_to_f32(unsigned short h) {
    return __uint_as_float(((unsigned int)h) << 16);
}

// Tensor Data Mover: DMA a 16-row x 256-col bf16 (2B) row-major tile from
// global memory into LDS.  D# per CDNA5 ISA ch.8 (2-D tensor, groups 2/3 = 0).
// This toolchain exposes the 6-argument builtin:
//   (uint32x4 g0, int32x8 g1, int32x4 g2, int32x4 g3, int32x8 extra, i32 cpol)
__device__ __forceinline__ void tdm_load_tile_16x256_bf16(
    const unsigned short* gtile, unsigned lds_byte_off)
{
    const unsigned long long ga = (unsigned long long)(uintptr_t)gtile;
    v4u g0;
    g0[0] = 1u;                                    // count=1, user-mode load
    g0[1] = lds_byte_off;                          // lds_addr (bytes)
    g0[2] = (unsigned)(ga & 0xFFFFFFFFu);          // global_addr[31:0]
    g0[3] = (unsigned)((ga >> 32) & 0x01FFFFFFu)   // global_addr[56:32]
            | (2u << 30);                          // type = 2 ("image")
    v8i g1;
    g1[0] = 0x00010000;    // workgroup_mask=0 (no cluster), data_size=1 (2 B)
    g1[1] = 0x01000000;    // atomic_barrier_addr=0 ; tensor_dim0[15:0]=256
    g1[2] = 0x04000000;    // tensor_dim0[31:16]=0 ; tensor_dim1[15:0]=1024
    g1[3] = 0x01000000;    // tensor_dim1[31:16]=0 ; tile_dim0=256
    g1[4] = 16;            // tile_dim1=16 rows ; tile_dim2=0
    g1[5] = 256;           // tensor_dim0_stride[31:0] = 256 elements
    g1[6] = 0;             // tensor_dim0_stride[47:32]=0 ; dim1_stride lo=0
    g1[7] = 0;             // tensor_dim1_stride hi = 0
    v4i g2 = {0, 0, 0, 0};                         // 2-D tile: groups 2/3 unused
    v4i g3 = {0, 0, 0, 0};
    v8i gx = {0, 0, 0, 0, 0, 0, 0, 0};             // extra group (unused)
    __builtin_amdgcn_tensor_load_to_lds(g0, g1, g2, g3, gx, 0);
}

// ---------------------------------------------------------------------------
// Kernel 1: codebook -> bf16 hi/lo split + per-code squared L2 norm (fp32).
// ---------------------------------------------------------------------------
__global__ __launch_bounds__(256) void vq_prep_weight(
    const float* __restrict__ w,
    unsigned short* __restrict__ whi,
    unsigned short* __restrict__ wlo,
    float* __restrict__ wnorm)
{
    const int code = blockIdx.x;
    const int c    = threadIdx.x;
    const float f  = w[code * CDIM + c];
    const unsigned short h = bf16_rne(f);
    whi[code * CDIM + c] = h;
    wlo[code * CDIM + c] = bf16_rne(f - bf16_to_f32(h));

    __shared__ float red[256];
    red[c] = f * f;
    __syncthreads();
    for (int s = 128; s > 0; s >>= 1) {
        if (c < s) red[c] += red[c + s];
        __syncthreads();
    }
    if (c == 0) wnorm[code] = red[0];
}

// ---------------------------------------------------------------------------
// Kernel 2: WMMA distance GEMM + per-row argmin, TDM-fed LDS double buffer.
// block = 256 threads = 8 waves; each wave owns 16 z-rows (A fragments in
// registers).  64 code tiles of 16; per tile 8 K-steps x 3 bf16 WMMAs.
// grid = N_VEC / 128 = 256 blocks.
//
// Fragment layouts (A and B share this code's contiguous k order):
//  A 16x32 bf16 : lane L holds row M=L&15 ; elems i=0..15 = K 32*kk+16*(L>=16)+i.
//  B 32x16 bf16 : lane L = column (code) L&15, K-half by L>=16, 16
//                 consecutive K ascending -> contiguous 32 B of the code row.
//  C/D          : lane L holds N=L&15 (code), VGPR j holds M = j + 8*(L>=16).
// ---------------------------------------------------------------------------
__global__ __launch_bounds__(256) void vq_argmin_wmma(
    const float* __restrict__ z,              // NCHW [32][256][1024]
    const unsigned short* __restrict__ whi,   // [1024][256] bf16 hi
    const unsigned short* __restrict__ wlo,   // [1024][256] bf16 lo
    const float* __restrict__ wnorm,          // [1024]
    int* __restrict__ minidx,                 // [N_VEC]
    float* __restrict__ outIdxF)              // d_out index region
{
    __shared__ unsigned short lds_bh[2][16 * CDIM];   // 8 KB per buffer
    __shared__ unsigned short lds_bl[2][16 * CDIM];   // 8 KB per buffer

    const int lane = threadIdx.x & 31;
    const int wv   = threadIdx.x >> 5;        // wave in block, 0..7
    const int col  = lane & 15;               // A-row select / B-column select
    const int half = lane >> 4;               // K-half select

    const int rowBase = blockIdx.x * 128 + wv * 16;
    const int r  = rowBase + col;             // this lane's z row (n index)
    const int b  = r >> 10;                   // batch
    const int hw = r & 1023;                  // h*W+w
    // z_flat[r][k] at z[(b*CDIM + k)*HWX + hw]; lanes 0..15 coalesce over hw.
    const float* zrow = z + (size_t)b * CDIM * HWX + hw;

    // Hoist all A fragments (hi/lo bf16 split of the fp32 row) into registers.
    v16bf ahi[8], alo[8];
#pragma unroll
    for (int kk = 0; kk < 8; ++kk) {
        const int k0 = kk * 32 + half * 16;
        v16u th, tl;
#pragma unroll
        for (int i = 0; i < 8; ++i) {
            const float f0 = zrow[(size_t)(k0 + i) * HWX];
            const float f1 = zrow[(size_t)(k0 + 8 + i) * HWX];
            const unsigned short h0 = bf16_rne(f0);
            const unsigned short h1 = bf16_rne(f1);
            th[i]     = h0;
            th[i + 8] = h1;
            tl[i]     = bf16_rne(f0 - bf16_to_f32(h0));
            tl[i + 8] = bf16_rne(f1 - bf16_to_f32(h1));
        }
        ahi[kk] = __builtin_bit_cast(v16bf, th);
        alo[kk] = __builtin_bit_cast(v16bf, tl);
    }

    float bestv[8];
    int   besti[8];
#pragma unroll
    for (int j = 0; j < 8; ++j) { bestv[j] = 3.4e38f; besti[j] = 0; }

    // Prologue: TDM tile 0 into buffer 0 (wave 0 issues; TENSORcnt tracks).
    if (wv == 0) {
        tdm_load_tile_16x256_bf16(whi, (unsigned)(uintptr_t)&lds_bh[0][0]);
        tdm_load_tile_16x256_bf16(wlo, (unsigned)(uintptr_t)&lds_bl[0][0]);
        __builtin_amdgcn_s_wait_tensorcnt(0);
    }
    __syncthreads();

    for (int t = 0; t < K_CODES / 16; ++t) {
        const int buf = t & 1;

        // Kick off DMA of tile t+1 into the other buffer.  The barrier at the
        // end of iteration t-1 guarantees no wave still reads that buffer.
        if (wv == 0 && t + 1 < K_CODES / 16) {
            const size_t toff = (size_t)(t + 1) * 16 * CDIM;
            tdm_load_tile_16x256_bf16(whi + toff,
                                      (unsigned)(uintptr_t)&lds_bh[buf ^ 1][0]);
            tdm_load_tile_16x256_bf16(wlo + toff,
                                      (unsigned)(uintptr_t)&lds_bl[buf ^ 1][0]);
        }

        const int code = t * 16 + col;
        const unsigned short* bh = &lds_bh[buf][col * CDIM + half * 16];
        const unsigned short* bl = &lds_bl[buf][col * CDIM + half * 16];

        v8f acc = {};
#pragma unroll
        for (int kk = 0; kk < 8; ++kk) {
            const v16bf bhi = __builtin_bit_cast(v16bf, *(const v16u*)(bh + kk * 32));
            const v16bf blo = __builtin_bit_cast(v16bf, *(const v16u*)(bl + kk * 32));
            // dot ~= zhi*whi + zhi*wlo + zlo*whi  (fp32 accumulate)
            acc = __builtin_amdgcn_wmma_f32_16x16x32_bf16(false, ahi[kk], false, bhi,
                                                          (short)0, acc, false, false);
            acc = __builtin_amdgcn_wmma_f32_16x16x32_bf16(false, ahi[kk], false, blo,
                                                          (short)0, acc, false, false);
            acc = __builtin_amdgcn_wmma_f32_16x16x32_bf16(false, alo[kk], false, bhi,
                                                          (short)0, acc, false, false);
        }

        const float wn = wnorm[code];
#pragma unroll
        for (int j = 0; j < 8; ++j) {
            const float d = wn - 2.0f * acc[j];
            if (d < bestv[j]) { bestv[j] = d; besti[j] = code; }  // strict < keeps first idx
        }

        // Tile t+1 must be fully in LDS before anyone touches it next round.
        if (wv == 0) __builtin_amdgcn_s_wait_tensorcnt(0);
        __syncthreads();
    }

    // Cross-lane argmin over the 16 codes held by lanes sharing a row-half
    // (xor offsets < 16 stay inside each 16-lane half).
#pragma unroll
    for (int off = 8; off > 0; off >>= 1) {
#pragma unroll
        for (int j = 0; j < 8; ++j) {
            const float ov = __shfl_xor(bestv[j], off, 32);
            const int   oi = __shfl_xor(besti[j], off, 32);
            if (ov < bestv[j] || (ov == bestv[j] && oi < besti[j])) {
                bestv[j] = ov; besti[j] = oi;
            }
        }
    }

    if (col == 0) {
#pragma unroll
        for (int j = 0; j < 8; ++j) {
            const int n = rowBase + j + 8 * half;   // C-layout: M = j + 8*half
            minidx[n]  = besti[j];
            outIdxF[n] = (float)besti[j];
        }
    }
}

// ---------------------------------------------------------------------------
// Kernel 3: gather z_q = weight[minidx], write straight-through output (NCHW,
// numerically == z_q), accumulate per-block partial sums of (z - z_q)^2.
// ---------------------------------------------------------------------------
__global__ __launch_bounds__(256) void vq_gather(
    const float* __restrict__ z,
    const float* __restrict__ w,
    const int* __restrict__ minidx,
    float* __restrict__ outq,
    float* __restrict__ partial)
{
    float s = 0.0f;
    for (long long e = (long long)blockIdx.x * blockDim.x + threadIdx.x;
         e < OUTQ_ELEMS;
         e += (long long)gridDim.x * blockDim.x) {
        const int bb  = (int)(e >> 18);          // / (C*HW)
        const int c   = (int)((e >> 10) & 255);
        const int hw  = (int)(e & 1023);
        const int n   = bb * HWX + hw;
        const float zq = w[minidx[n] * CDIM + c];
        const float d  = z[e] - zq;
        s += d * d;
        outq[e] = zq;                            // z + sg(z_q - z) == z_q
    }
    __shared__ float red[256];
    red[threadIdx.x] = s;
    __syncthreads();
    for (int t = 128; t > 0; t >>= 1) {
        if (threadIdx.x < t) red[threadIdx.x] += red[threadIdx.x + t];
        __syncthreads();
    }
    if (threadIdx.x == 0) partial[blockIdx.x] = red[0];
}

// ---------------------------------------------------------------------------
// Kernel 4: deterministic final loss reduction.
// loss = mean((sg(z)-zq)^2) + beta*mean((z-sg(zq))^2) = (1+beta)*mean(diff^2)
// ---------------------------------------------------------------------------
__global__ __launch_bounds__(256) void vq_loss(
    const float* __restrict__ partial, int np, float* __restrict__ lossOut)
{
    __shared__ float red[256];
    float s = 0.0f;
    for (int i = threadIdx.x; i < np; i += 256) s += partial[i];
    red[threadIdx.x] = s;
    __syncthreads();
    for (int t = 128; t > 0; t >>= 1) {
        if (threadIdx.x < t) red[threadIdx.x] += red[threadIdx.x + t];
        __syncthreads();
    }
    if (threadIdx.x == 0)
        *lossOut = red[0] * (1.0f + BETA_F) / (float)OUTQ_ELEMS;
}

// ---------------------------------------------------------------------------
extern "C" void kernel_launch(void* const* d_in, const int* in_sizes, int n_in,
                              void* d_out, int out_size, void* d_ws, size_t ws_size,
                              hipStream_t stream) {
    const float* z = (const float*)d_in[0];   // [32,256,32,32] fp32
    const float* w = (const float*)d_in[1];   // [1024,256] fp32
    float* out = (float*)d_out;

    // Workspace layout (~1.2 MB total)
    char* ws = (char*)d_ws;
    unsigned short* whi   = (unsigned short*)(ws);                         // 512 KB
    unsigned short* wlo   = (unsigned short*)(ws + (512 << 10));           // 512 KB
    float*          wnorm = (float*)(ws + (1024 << 10));                   //   4 KB
    int*            midx  = (int*)(ws + (1024 << 10) + (4 << 10));         // 128 KB
    float*          part  = (float*)(ws + (1024 << 10) + (132 << 10));     //   4 KB

    // d_out layout: [z_q_out 8388608][loss 1][min_indices 32768]
    float* outq    = out;
    float* lossOut = out + OUTQ_ELEMS;
    float* outIdxF = out + OUTQ_ELEMS + 1;

    vq_prep_weight<<<K_CODES, 256, 0, stream>>>(w, whi, wlo, wnorm);
    vq_argmin_wmma<<<N_VEC / 128, 256, 0, stream>>>(z, whi, wlo, wnorm, midx, outIdxF);
    const int NB = 1024;
    vq_gather<<<NB, 256, 0, stream>>>(z, w, midx, outq, part);
    vq_loss<<<1, 256, 0, stream>>>(part, NB, lossOut);
}
